// MultiHeadAttentionLayer_65884798321389
// MI455X (gfx1250) — matrix operand order repaired and checked by reading, not compiled, not run on hardware
//
#include <hip/hip_runtime.h>
#include <math.h>

typedef __attribute__((ext_vector_type(16))) _Float16     v16h;
typedef __attribute__((ext_vector_type(8)))  float        v8f;
typedef __attribute__((ext_vector_type(4)))  unsigned int u32x4;
typedef __attribute__((ext_vector_type(4)))  int          v4i;

#if __has_builtin(__builtin_amdgcn_global_load_async_to_lds_b128) && \
    __has_builtin(__builtin_amdgcn_s_wait_asynccnt)
#define USE_ASYNC_LDS 1
// Builtin signature (from hipcc diagnostics): arg0 = addrspace(1) v4i*,
// arg1 = LDS-side v4i*, arg2/arg3 = imm offset / cpol.
typedef __attribute__((address_space(1))) v4i g_v4i;
typedef __attribute__((address_space(3))) v4i l_v4i;
#endif

namespace {
constexpr int    B_ = 2, L_ = 4096, D_ = 128, H_ = 4;
constexpr size_t BHLD_ = (size_t)B_ * H_ * L_ * D_;   // 4,194,304 elements
constexpr size_t XSZ_  = (size_t)B_ * L_ * D_;        // 1,048,576
constexpr size_t WSZ_  = (size_t)H_ * D_ * D_;        // 65,536 (per Wq/Wk/Wv/Wo)
}

// POD fragment: 16 halves per lane = 8 VGPRs (A or B operand of 16x16x32 f16 WMMA)
union Frag {
  v16h     v;
  u32x4    q[2];
  _Float16 h[16];
};

__device__ __forceinline__ v8f wmma16(const Frag& a, const Frag& b, v8f c) {
  // D = A(16x32 f16) x B(32x16 f16) + C(16x16 f32)
  return __builtin_amdgcn_wmma_f32_16x16x32_f16(false, a.v, false, b.v,
                                                (short)0, c, false, false);
}

// ---------------------------------------------------------------------------
// Kernel 0: one-shot f32 -> f16 conversion (x and all weights), so the GEMM
// kernels do pure vector f16 loads (no per-block v_cvt spam).
// ---------------------------------------------------------------------------
__global__ void __launch_bounds__(256) cvt_f16_kernel(
    const float* __restrict__ s, _Float16* __restrict__ d, int n)
{
  const int i = (blockIdx.x * 256 + threadIdx.x) * 4;
  if (i + 3 < n) {
    #pragma unroll
    for (int t = 0; t < 4; ++t) d[i + t] = (_Float16)s[i + t];
  }
}

// ---------------------------------------------------------------------------
// Kernel 1: q/k/v = x @ W^T per head, written as f16 [b,h,l,e].
// grid = (B*L/128, 12)  y: proj = y>>2 (0=q,1=k,2=v), h = y&3. block = 256.
// ---------------------------------------------------------------------------
__global__ void __launch_bounds__(256) qkv_proj_kernel(
    const _Float16* __restrict__ xh,  const _Float16* __restrict__ wq,
    const _Float16* __restrict__ wk,  const _Float16* __restrict__ wv,
    _Float16* __restrict__ ws)
{
  const int lane = threadIdx.x & 31;
  const int wave = threadIdx.x >> 5;
  const int r    = lane & 15;
  const int hi   = lane >> 4;
  const int pj   = blockIdx.y >> 2;
  const int h    = blockIdx.y & 3;
  const _Float16* W = (pj == 0) ? wq : ((pj == 1) ? wk : wv);
  _Float16* out  = ws + (size_t)pj * BHLD_;

  const int grow0 = blockIdx.x * 128 + wave * 16;      // global row (b*L + l)
  const int bb    = grow0 / L_;                        // block stays in one b

  // A fragments: x[row, 0..127] -> 4 K-blocks of 32
  Frag a[4];
  const _Float16* xrow = xh + (size_t)(grow0 + r) * D_;
  #pragma unroll
  for (int kb = 0; kb < 4; ++kb) {
    a[kb].q[0] = *(const u32x4*)(xrow + kb * 32 + hi * 16);
    a[kb].q[1] = *(const u32x4*)(xrow + kb * 32 + hi * 16 + 8);
  }

  for (int nt = 0; nt < 8; ++nt) {
    // B[k][n] = W[h*128 + nt*16 + n][k]; lane n=r reads 16 contiguous halves
    const _Float16* wrow = W + (size_t)(h * D_ + nt * 16 + r) * D_;
    v8f c = {0.f, 0.f, 0.f, 0.f, 0.f, 0.f, 0.f, 0.f};
    #pragma unroll
    for (int kb = 0; kb < 4; ++kb) {
      Frag bf;
      bf.q[0] = *(const u32x4*)(wrow + kb * 32 + hi * 16);
      bf.q[1] = *(const u32x4*)(wrow + kb * 32 + hi * 16 + 8);
      c = wmma16(a[kb], bf, c);
    }
    // C element (row = i + 8*hi, col = nt*16 + r)
    #pragma unroll
    for (int i = 0; i < 8; ++i) {
      const int l = (grow0 + i + 8 * hi) % L_;
      out[(((size_t)bb * H_ + h) * L_ + l) * D_ + (nt * 16 + r)] = (_Float16)c[i];
    }
  }
}

// ---------------------------------------------------------------------------
// Kernel 2: flash attention. grid = (L/128, B*H), block = 256 (8 waves).
// Each wave owns 16 query rows; K/V streamed through LDS in 32-key tiles.
// K tile uses GLOBAL_LOAD_ASYNC_TO_LDS_B128 (ASYNCcnt) when available.
// ---------------------------------------------------------------------------
__global__ void __launch_bounds__(256) flash_attn_kernel(
    const _Float16* __restrict__ ws, _Float16* __restrict__ ctxo)
{
  __shared__ __align__(16) _Float16 lK [32][136];   // [key][d], padded stride
  __shared__ __align__(16) _Float16 lVt[128][40];   // [d][key], transposed
  __shared__ __align__(16) _Float16 lP [8][16][40]; // per-wave P tile

  const int lane  = threadIdx.x & 31;
  const int wave  = threadIdx.x >> 5;
  const int r     = lane & 15;
  const int hi    = lane >> 4;
  const int b     = blockIdx.y >> 2;
  const int h     = blockIdx.y & 3;
  const int m0    = blockIdx.x * 128;
  const int wrow0 = m0 + wave * 16;
  const size_t base = (((size_t)b * H_ + h) * L_) * (size_t)D_;
  const _Float16* qp = ws;
  const _Float16* kp = ws + BHLD_;
  const _Float16* vp = ws + 2 * BHLD_;

  // Q fragments for this wave's 16 rows (D=128 -> 4 K-blocks)
  Frag qf[4];
  {
    const _Float16* qrow = qp + base + (size_t)(wrow0 + r) * D_;
    #pragma unroll
    for (int kb = 0; kb < 4; ++kb) {
      qf[kb].q[0] = *(const u32x4*)(qrow + kb * 32 + hi * 16);
      qf[kb].q[1] = *(const u32x4*)(qrow + kb * 32 + hi * 16 + 8);
    }
  }

  const v8f vzero = {0.f, 0.f, 0.f, 0.f, 0.f, 0.f, 0.f, 0.f};
  v8f acc[8];
  #pragma unroll
  for (int d = 0; d < 8; ++d) acc[d] = vzero;
  float mrow[8], lrow[8];
  #pragma unroll
  for (int i = 0; i < 8; ++i) { mrow[i] = -1.0e30f; lrow[i] = 0.f; }

  const int nKB   = (m0 + 128) / 32;        // causal: keys only up to m0+127
  const int krow  = threadIdx.x >> 3;       // 0..31  (staging row)
  const int cbase = (threadIdx.x & 7) * 16; // 0..112 (staging col base)

  for (int kb0 = 0; kb0 < nKB; ++kb0) {
    const int k0 = kb0 * 32;
    __syncthreads();  // previous tile fully consumed
    {
      const _Float16* gk = kp + base + (size_t)(k0 + krow) * D_ + cbase;
#if defined(USE_ASYNC_LDS)
      // K: async DMA straight into LDS, no VGPR round trip (ASYNCcnt path)
      __builtin_amdgcn_global_load_async_to_lds_b128(
          (g_v4i*)gk, (l_v4i*)&lK[krow][cbase], 0, 0);
      __builtin_amdgcn_global_load_async_to_lds_b128(
          (g_v4i*)gk, (l_v4i*)&lK[krow][cbase], 16, 0);
#else
      *(u32x4*)&lK[krow][cbase]     = *(const u32x4*)gk;
      *(u32x4*)&lK[krow][cbase + 8] = *(const u32x4*)(gk + 8);
#endif
      // V: load + transpose into lVt (DMA cannot transpose)
      const _Float16* gv = vp + base + (size_t)(k0 + krow) * D_ + cbase;
      Frag tv;
      tv.q[0] = *(const u32x4*)gv;
      tv.q[1] = *(const u32x4*)(gv + 8);
      #pragma unroll
      for (int j = 0; j < 16; ++j) lVt[cbase + j][krow] = tv.h[j];
      if (kb0 + 1 < nKB) {                 // gfx1250 global_prefetch_b8
        __builtin_prefetch(gk + 32 * D_, 0, 1);
        __builtin_prefetch(gv + 32 * D_, 0, 1);
      }
#if defined(USE_ASYNC_LDS)
      __builtin_amdgcn_s_wait_asynccnt(0); // K tile landed in LDS
#endif
    }
    __syncthreads();  // tile visible to all waves

    if (k0 > wrow0 + 15) continue;  // whole tile above the causal diagonal

    // ---- S = Q * K^T : two 16x16 tiles over 32 keys -----------------------
    v8f s0 = vzero, s1 = vzero;
    {
      const _Float16* kr0 = &lK[r][0];
      const _Float16* kr1 = &lK[16 + r][0];
      #pragma unroll
      for (int kk = 0; kk < 4; ++kk) {
        Frag bf0, bf1;
        bf0.q[0] = *(const u32x4*)(kr0 + kk * 32 + hi * 16);
        bf0.q[1] = *(const u32x4*)(kr0 + kk * 32 + hi * 16 + 8);
        bf1.q[0] = *(const u32x4*)(kr1 + kk * 32 + hi * 16);
        bf1.q[1] = *(const u32x4*)(kr1 + kk * 32 + hi * 16 + 8);
        s0 = wmma16(qf[kk], bf0, s0);
        s1 = wmma16(qf[kk], bf1, s1);
      }
    }
    // ---- causal mask ------------------------------------------------------
    if (k0 + 31 > wrow0) {
      #pragma unroll
      for (int i = 0; i < 8; ++i) {
        const int rowg = wrow0 + i + 8 * hi;
        if (k0 + r > rowg)      s0[i] = -1.0e30f;
        if (k0 + 16 + r > rowg) s1[i] = -1.0e30f;
      }
    }
    // ---- online softmax (rows live across 16 lanes of each half-wave) -----
    float pmax[8];
    #pragma unroll
    for (int i = 0; i < 8; ++i) pmax[i] = fmaxf(s0[i], s1[i]);
    #pragma unroll
    for (int off = 1; off < 16; off <<= 1) {
      #pragma unroll
      for (int i = 0; i < 8; ++i)
        pmax[i] = fmaxf(pmax[i], __shfl_xor(pmax[i], off, 32));
    }
    float scl[8], rsum[8];
    #pragma unroll
    for (int i = 0; i < 8; ++i) {
      const float mn = fmaxf(mrow[i], pmax[i]);
      scl[i]  = __expf(mrow[i] - mn);
      mrow[i] = mn;
      const float p0 = __expf(s0[i] - mn);
      const float p1 = __expf(s1[i] - mn);
      s0[i] = p0; s1[i] = p1;
      rsum[i] = p0 + p1;
    }
    #pragma unroll
    for (int off = 1; off < 16; off <<= 1) {
      #pragma unroll
      for (int i = 0; i < 8; ++i) rsum[i] += __shfl_xor(rsum[i], off, 32);
    }
    #pragma unroll
    for (int i = 0; i < 8; ++i) lrow[i] = lrow[i] * scl[i] + rsum[i];
    #pragma unroll
    for (int d = 0; d < 8; ++d) {
      #pragma unroll
      for (int i = 0; i < 8; ++i) acc[d][i] *= scl[i];
    }
    // ---- P: C-layout -> LDS -> A-layout ----------------------------------
    #pragma unroll
    for (int i = 0; i < 8; ++i) {
      lP[wave][i + 8 * hi][r]      = (_Float16)s0[i];
      lP[wave][i + 8 * hi][16 + r] = (_Float16)s1[i];
    }
    Frag pf;  // same-wave LDS RAW: compiler inserts s_wait_dscnt
    {
      const _Float16* prow = &lP[wave][r][0];
      pf.q[0] = *(const u32x4*)(prow + hi * 16);
      pf.q[1] = *(const u32x4*)(prow + hi * 16 + 8);
    }
    // ---- ctx += P * V : 8 dim-tiles, K = 32 keys --------------------------
    #pragma unroll
    for (int d = 0; d < 8; ++d) {
      Frag bf;
      const _Float16* vtr = &lVt[d * 16 + r][0];
      bf.q[0] = *(const u32x4*)(vtr + hi * 16);
      bf.q[1] = *(const u32x4*)(vtr + hi * 16 + 8);
      acc[d] = wmma16(pf, bf, acc[d]);
    }
  }

  // normalize (one rcp per row), write ctx as f16 [b, l, h, e]
  float rinv[8];
  #pragma unroll
  for (int i = 0; i < 8; ++i) rinv[i] = 1.0f / lrow[i];
  #pragma unroll
  for (int d = 0; d < 8; ++d) {
    #pragma unroll
    for (int i = 0; i < 8; ++i) {
      const int l = wrow0 + i + 8 * hi;
      ctxo[(((size_t)b * L_ + l) * H_ + h) * D_ + d * 16 + r] =
          (_Float16)(acc[d][i] * rinv[i]);
    }
  }
}

// ---------------------------------------------------------------------------
// Kernel 3: y = ctx @ Wo^T.  ctx f16 [B*L, 512], Wo f16 [128, 512], y f32.
// grid = B*L/128, block = 256.
// ---------------------------------------------------------------------------
__global__ void __launch_bounds__(256) out_proj_kernel(
    const _Float16* __restrict__ ctx, const _Float16* __restrict__ wo,
    float* __restrict__ y)
{
  const int lane  = threadIdx.x & 31;
  const int wave  = threadIdx.x >> 5;
  const int r     = lane & 15;
  const int hi    = lane >> 4;
  const int grow0 = blockIdx.x * 128 + wave * 16;

  Frag a[16];   // K = 512 -> 16 K-blocks of 32
  {
    const _Float16* crow = ctx + (size_t)(grow0 + r) * (H_ * D_);
    #pragma unroll
    for (int kb = 0; kb < 16; ++kb) {
      a[kb].q[0] = *(const u32x4*)(crow + kb * 32 + hi * 16);
      a[kb].q[1] = *(const u32x4*)(crow + kb * 32 + hi * 16 + 8);
    }
  }
  for (int nt = 0; nt < 8; ++nt) {
    const _Float16* wrow = wo + (size_t)(nt * 16 + r) * (H_ * D_);
    v8f c = {0.f, 0.f, 0.f, 0.f, 0.f, 0.f, 0.f, 0.f};
    #pragma unroll
    for (int kb = 0; kb < 16; ++kb) {
      Frag bf;
      bf.q[0] = *(const u32x4*)(wrow + kb * 32 + hi * 16);
      bf.q[1] = *(const u32x4*)(wrow + kb * 32 + hi * 16 + 8);
      c = wmma16(a[kb], bf, c);
    }
    #pragma unroll
    for (int i = 0; i < 8; ++i)
      y[(size_t)(grow0 + i + 8 * hi) * D_ + nt * 16 + r] = c[i];
  }
}

// ---------------------------------------------------------------------------
extern "C" void kernel_launch(void* const* d_in, const int* in_sizes, int n_in,
                              void* d_out, int out_size, void* d_ws, size_t ws_size,
                              hipStream_t stream) {
  const float* x  = (const float*)d_in[0];
  const float* Wq = (const float*)d_in[1];
  const float* Wk = (const float*)d_in[2];
  const float* Wv = (const float*)d_in[3];
  const float* Wo = (const float*)d_in[4];

  // workspace layout (f16 elements):
  //   q | k | v | ctx            (4 * 4M = 32 MB)
  //   xh | Wqh | Wkh | Wvh | Woh (~2.6 MB)
  _Float16* ws  = (_Float16*)d_ws;
  _Float16* ctx = ws + 3 * BHLD_;
  _Float16* xh  = ws + 4 * BHLD_;
  _Float16* wqh = xh  + XSZ_;
  _Float16* wkh = wqh + WSZ_;
  _Float16* wvh = wkh + WSZ_;
  _Float16* woh = wvh + WSZ_;

  const int nx = (int)XSZ_, nw = (int)WSZ_;
  cvt_f16_kernel<<<dim3((nx / 4 + 255) / 256), 256, 0, stream>>>(x,  xh,  nx);
  cvt_f16_kernel<<<dim3((nw / 4 + 255) / 256), 256, 0, stream>>>(Wq, wqh, nw);
  cvt_f16_kernel<<<dim3((nw / 4 + 255) / 256), 256, 0, stream>>>(Wk, wkh, nw);
  cvt_f16_kernel<<<dim3((nw / 4 + 255) / 256), 256, 0, stream>>>(Wv, wvh, nw);
  cvt_f16_kernel<<<dim3((nw / 4 + 255) / 256), 256, 0, stream>>>(Wo, woh, nw);

  qkv_proj_kernel<<<dim3((B_ * L_) / 128, 12), 256, 0, stream>>>(xh, wqh, wkh, wvh, ws);
  flash_attn_kernel<<<dim3(L_ / 128, B_ * H_), 256, 0, stream>>>(ws, ctx);
  out_proj_kernel<<<dim3((B_ * L_) / 128), 256, 0, stream>>>(ctx, woh, (float*)d_out);

  (void)in_sizes; (void)n_in; (void)out_size; (void)ws_size;
}
